// MultiHeadAttention_21534966022849
// MI455X (gfx1250) — matrix operand order repaired and checked by reading, not compiled, not run on hardware
//
#include <hip/hip_runtime.h>
#include <hip/hip_bf16.h>
#include <stdint.h>

// Problem constants (from reference)
#define Bc   8
#define Nc   2048
#define Dc   512
#define Hc   8
#define DHc  64
#define SCALEc 0.125f   // DH^-0.5
#define NKB  (Nc / 64)  // key blocks of 64

typedef __attribute__((ext_vector_type(16))) __bf16 v16bf;
typedef __attribute__((ext_vector_type(8)))  float  v8f;

union FragBF { v16bf v; unsigned short u[16]; };

__device__ __forceinline__ unsigned short f2bf(float f) {
  union { float f; uint32_t u; } x; x.f = f;
  uint32_t r = x.u + 0x7FFFu + ((x.u >> 16) & 1u);   // round-to-nearest-even
  return (unsigned short)(r >> 16);
}

__device__ __forceinline__ v8f wmma_bf16(v16bf a, v16bf b, v8f c) {
  // (neg_a, A, neg_b, B, c_mod, C, reuse_a, reuse_b)
  return __builtin_amdgcn_wmma_f32_16x16x32_bf16(false, a, false, b, (short)0, c,
                                                 false, false);
}

// ---- CDNA5 async global->LDS copy via inline asm (ASYNCcnt) ----------------
// GLOBAL_LOAD_ASYNC_TO_LDS_B128, GVS mode: mem = SADDR(u64) + VADDR(u32 bytes);
// VDST = per-lane LDS byte address (low 32 bits of generic LDS pointer).
__device__ __forceinline__ void async_copy_b128(void* lds, unsigned voff_bytes,
                                                const void* sbase) {
  unsigned lds_off = (unsigned)(uintptr_t)lds;
  asm volatile("global_load_async_to_lds_b128 %0, %1, %2"
               :
               : "v"(lds_off), "v"(voff_bytes),
                 "s"((unsigned long long)(uintptr_t)sbase)
               : "memory");
}

__device__ __forceinline__ void async_wait0() {
#if __has_builtin(__builtin_amdgcn_s_wait_asynccnt)
  __builtin_amdgcn_s_wait_asynccnt(0);
#else
  asm volatile("s_wait_asynccnt 0" ::: "memory");
#endif
}

// ---------------------------------------------------------------------------
// Kernel 1: qkv = x[16384,512] @ W_qkv[512,1536]  -> Q/K/V bf16 [B,H,N,DH]
// 64x64 tile per 256-thread block (8 waves, each wave: 16 rows x 32 cols)
// ---------------------------------------------------------------------------
__global__ __launch_bounds__(256) void qkv_kernel(
    const float* __restrict__ x, const float* __restrict__ Wqkv,
    unsigned short* __restrict__ Qb, unsigned short* __restrict__ Kb,
    unsigned short* __restrict__ Vb)
{
  __shared__ unsigned short As[64][36];   // [m][k], padded
  __shared__ unsigned short Bt[64][36];   // [n][k], padded (W transposed)
  const int t    = threadIdx.x;
  const int lane = t & 31, wave = t >> 5;
  const int l15  = lane & 15, lhi = lane >> 4;
  const int row0 = blockIdx.x * 64;
  const int col0 = blockIdx.y * 64;
  const int mr   = (wave & 3) * 16;
  const int nc   = (wave >> 2) * 32;

  v8f acc0 = {}; v8f acc1 = {};
  for (int k0 = 0; k0 < Dc; k0 += 32) {
    __syncthreads();
#pragma unroll
    for (int i = 0; i < 8; ++i) {          // stage A: 64x32 fp32 -> bf16
      int e = t * 8 + i;
      int m = e >> 5, k = e & 31;
      As[m][k] = f2bf(x[(size_t)(row0 + m) * Dc + k0 + k]);
    }
#pragma unroll
    for (int i = 0; i < 8; ++i) {          // stage B transposed: 32x64 -> Bt[n][k]
      int e = t * 8 + i;
      int k = e >> 6, n = e & 63;
      Bt[n][k] = f2bf(Wqkv[(size_t)(k0 + k) * (3 * Dc) + col0 + n]);
    }
    __syncthreads();

    FragBF a, b0, b1;
#pragma unroll
    for (int e = 0; e < 16; ++e) {         // A frag: k = e + 16*laneHi
      int kk = e + 16 * lhi;
      a.u[e] = As[mr + l15][kk];
    }
#pragma unroll
    for (int e = 0; e < 16; ++e) {         // B frag: k = e + 16*laneHi
      int kk = e + 16 * lhi;
      b0.u[e] = Bt[nc + l15][kk];
      b1.u[e] = Bt[nc + 16 + l15][kk];
    }
    acc0 = wmma_bf16(a.v, b0.v, acc0);
    acc1 = wmma_bf16(a.v, b1.v, acc1);
  }

  // scatter: col c -> (s,h,dh); row -> (b, nseq); layout [B,H,N,DH]
#pragma unroll
  for (int r = 0; r < 8; ++r) {
    int m  = mr + r + 8 * lhi;
    int rg = row0 + m;
    int bb = rg >> 11;            // / N
    int ns = rg & (Nc - 1);
#pragma unroll
    for (int half = 0; half < 2; ++half) {
      int c  = col0 + nc + half * 16 + l15;
      int s  = c >> 9;            // 0=Q,1=K,2=V
      int hh = (c >> 6) & 7;
      int dh = c & 63;
      unsigned short* dst = (s == 0) ? Qb : ((s == 1) ? Kb : Vb);
      float val = (half == 0) ? acc0[r] : acc1[r];
      dst[(((size_t)bb * Hc + hh) * Nc + ns) * DHc + dh] = f2bf(val);
    }
  }
}

// ---------------------------------------------------------------------------
// Kernel 2: flash attention per (b, h, 128 q-rows). 8 waves, one 16-row
// strip per wave; online softmax over 32 key-blocks of 64.
// Double-buffered async (ASYNCcnt) K/V staging overlaps HBM with WMMA.
// ---------------------------------------------------------------------------
__global__ __launch_bounds__(256) void attn_kernel(
    const unsigned short* __restrict__ Qb, const unsigned short* __restrict__ Kb,
    const unsigned short* __restrict__ Vb, const float* __restrict__ bias,
    const int* __restrict__ mask, unsigned short* __restrict__ Ob)
{
  __shared__ unsigned short Ks[2][64][72];    // [buf][key][dh] (pitch 144B, 16B-mult)
  __shared__ unsigned short Vs[2][64][72];    // [buf][key][dh]
  __shared__ unsigned short Ps[8][16][72];    // per-wave P relayout scratch

  const int t    = threadIdx.x;
  const int lane = t & 31, wave = t >> 5;
  const int l15  = lane & 15, lhi = lane >> 4;
  const int bb   = blockIdx.z, hh = blockIdx.y;
  const int qs   = blockIdx.x * 128 + wave * 16;   // wave strip base (q row)

  const size_t headBase = ((size_t)bb * Hc + hh) * (size_t)Nc * DHc;
  const unsigned short* Kh = Kb + headBase;
  const unsigned short* Vh = Vb + headBase;

  // issue async stage of one 64x64 K tile + V tile into buffer `buf`
  auto stage = [&](int kb, int buf) {
#pragma unroll
    for (int i = 0; i < 2; ++i) {
      int c   = t * 2 + i;            // 512 x 16B chunks per tile
      int row = c >> 3;
      int cs  = (c & 7) * 8;          // ushort offset, 16B chunk
      unsigned voff = (unsigned)(((kb * 64 + row) * DHc + cs) * 2);   // bytes
      async_copy_b128(&Ks[buf][row][cs], voff, Kh);
      async_copy_b128(&Vs[buf][row][cs], voff, Vh);
    }
  };

  // persistent Q fragments (DH=64 -> 2 k-steps of 32)
  FragBF qf[2];
  {
    const unsigned short* Qrow = Qb + headBase + (size_t)(qs + l15) * DHc;
#pragma unroll
    for (int ks = 0; ks < 2; ++ks)
#pragma unroll
      for (int e = 0; e < 16; ++e) {
        int kk = ks * 32 + e + 16 * lhi;
        qf[ks].u[e] = Qrow[kk];
      }
  }

  v8f o[4] = {};                 // O strip 16x64 fp32
  float m_r[8], l_r[8];
#pragma unroll
  for (int r = 0; r < 8; ++r) { m_r[r] = -1e30f; l_r[r] = 0.f; }

  stage(0, 0);                   // prologue fill

  for (int kb = 0; kb < NKB; ++kb) {
    const int buf = kb & 1;
    async_wait0();               // this wave's chunks for `buf` are in LDS
    __syncthreads();             // everyone's chunks visible; buf^1 free to fill

    if (kb + 1 < NKB) {
      stage(kb + 1, buf ^ 1);    // overlap next tile's HBM fetch with compute
      // prefetch next block's bias/mask stream (dominant HBM traffic):
      // one 128B line per lane covers this wave's 16x64 fp32 tile
      int prow = qs + (lane >> 1);
      int pcol = (kb + 1) * 64 + (lane & 1) * 32;
      size_t poff = ((size_t)bb * Nc + prow) * (size_t)Nc + pcol;
      __builtin_prefetch(&bias[poff], 0, 3);
      __builtin_prefetch(&mask[poff], 0, 3);
    }

    // S = Q * K^T  (B frag reads Ks[key][dh] contiguously)
    v8f s[4] = {};
#pragma unroll
    for (int ks = 0; ks < 2; ++ks) {
#pragma unroll
      for (int tc = 0; tc < 4; ++tc) {
        FragBF bf;
#pragma unroll
        for (int e = 0; e < 16; ++e)
          bf.u[e] = Ks[buf][tc * 16 + l15][ks * 32 + e + 16 * lhi];
        s[tc] = wmma_bf16(qf[ks].v, bf.v, s[tc]);
      }
    }

    // scale, mask (==1 -> -inf), + bias
#pragma unroll
    for (int tc = 0; tc < 4; ++tc)
#pragma unroll
      for (int r = 0; r < 8; ++r) {
        int qg = qs + r + 8 * lhi;
        int kg = kb * 64 + tc * 16 + l15;
        size_t off = ((size_t)bb * Nc + qg) * (size_t)Nc + kg;
        float v = s[tc][r] * SCALEc;
        v = (mask[off] == 1) ? -1e30f : (v + bias[off]);
        s[tc][r] = v;
      }

    // online softmax: rows r+8*lhi live in one 16-lane half -> xor<16 reduce
#pragma unroll
    for (int r = 0; r < 8; ++r) {
      float mx = fmaxf(fmaxf(s[0][r], s[1][r]), fmaxf(s[2][r], s[3][r]));
#pragma unroll
      for (int off = 1; off < 16; off <<= 1)
        mx = fmaxf(mx, __shfl_xor(mx, off, 32));
      float mn    = fmaxf(m_r[r], mx);
      float alpha = __expf(m_r[r] - mn);
      float psum  = 0.f;
#pragma unroll
      for (int tc = 0; tc < 4; ++tc) {
        float p = __expf(s[tc][r] - mn);
        s[tc][r] = p;
        psum += p;
      }
#pragma unroll
      for (int off = 1; off < 16; off <<= 1)
        psum += __shfl_xor(psum, off, 32);
      m_r[r] = mn;
      l_r[r] = l_r[r] * alpha + psum;
      o[0][r] *= alpha; o[1][r] *= alpha; o[2][r] *= alpha; o[3][r] *= alpha;
    }

    // relayout P (C-layout -> A-frag layout) through per-wave LDS scratch
#pragma unroll
    for (int tc = 0; tc < 4; ++tc)
#pragma unroll
      for (int r = 0; r < 8; ++r)
        Ps[wave][r + 8 * lhi][tc * 16 + l15] = f2bf(s[tc][r]);
    asm volatile("s_wait_dscnt 0" ::: "memory");   // intra-wave DS ordering

    FragBF pa[2];
#pragma unroll
    for (int ks = 0; ks < 2; ++ks)
#pragma unroll
      for (int e = 0; e < 16; ++e) {
        int kk = ks * 32 + e + 16 * lhi;
        pa[ks].u[e] = Ps[wave][l15][kk];
      }

    // O += P * V   (B frag: lane n = dh, k = key -> strided column read of Vs)
#pragma unroll
    for (int ks = 0; ks < 2; ++ks) {
#pragma unroll
      for (int td = 0; td < 4; ++td) {
        FragBF bf;
#pragma unroll
        for (int e = 0; e < 16; ++e)
          bf.u[e] = Vs[buf][ks * 32 + e + 16 * lhi][td * 16 + l15];
        o[td] = wmma_bf16(pa[ks].v, bf.v, o[td]);
      }
    }
  }

  // normalize and store concat-head layout [B, N, H*DH] in bf16
#pragma unroll
  for (int td = 0; td < 4; ++td)
#pragma unroll
    for (int r = 0; r < 8; ++r) {
      int qg = qs + r + 8 * lhi;
      int dh = td * 16 + l15;
      float val = o[td][r] / l_r[r];
      Ob[((size_t)bb * Nc + qg) * Dc + hh * DHc + dh] = f2bf(val);
    }
}

// ---------------------------------------------------------------------------
// Kernel 3: out = O[16384,512] @ W_proj[512,512], fp32 output
// ---------------------------------------------------------------------------
__global__ __launch_bounds__(256) void proj_kernel(
    const unsigned short* __restrict__ Ob, const float* __restrict__ Wp,
    float* __restrict__ out)
{
  __shared__ unsigned short As[64][36];
  __shared__ unsigned short Bt[64][36];
  const int t    = threadIdx.x;
  const int lane = t & 31, wave = t >> 5;
  const int l15  = lane & 15, lhi = lane >> 4;
  const int row0 = blockIdx.x * 64;
  const int col0 = blockIdx.y * 64;
  const int mr   = (wave & 3) * 16;
  const int nc   = (wave >> 2) * 32;

  v8f acc0 = {}; v8f acc1 = {};
  for (int k0 = 0; k0 < Dc; k0 += 32) {
    __syncthreads();
#pragma unroll
    for (int i = 0; i < 8; ++i) {
      int e = t * 8 + i;
      int m = e >> 5, k = e & 31;
      As[m][k] = Ob[(size_t)(row0 + m) * Dc + k0 + k];   // already bf16
    }
#pragma unroll
    for (int i = 0; i < 8; ++i) {
      int e = t * 8 + i;
      int k = e >> 6, n = e & 63;
      Bt[n][k] = f2bf(Wp[(size_t)(k0 + k) * Dc + col0 + n]);
    }
    __syncthreads();

    FragBF a, b0, b1;
#pragma unroll
    for (int e = 0; e < 16; ++e) {
      int kk = e + 16 * lhi;
      a.u[e] = As[mr + l15][kk];
    }
#pragma unroll
    for (int e = 0; e < 16; ++e) {
      int kk = e + 16 * lhi;
      b0.u[e] = Bt[nc + l15][kk];
      b1.u[e] = Bt[nc + 16 + l15][kk];
    }
    acc0 = wmma_bf16(a.v, b0.v, acc0);
    acc1 = wmma_bf16(a.v, b1.v, acc1);
  }

#pragma unroll
  for (int r = 0; r < 8; ++r) {
    int rg = row0 + mr + r + 8 * lhi;
    out[(size_t)rg * Dc + col0 + nc + l15]      = acc0[r];
    out[(size_t)rg * Dc + col0 + nc + 16 + l15] = acc1[r];
  }
}

// ---------------------------------------------------------------------------
extern "C" void kernel_launch(void* const* d_in, const int* in_sizes, int n_in,
                              void* d_out, int out_size, void* d_ws, size_t ws_size,
                              hipStream_t stream) {
  const float* x    = (const float*)d_in[0];   // [B,N,D]
  const float* bias = (const float*)d_in[1];   // [B,N,N]
  const int*   mask = (const int*)d_in[2];     // [B,1,N,N]
  const float* Wqkv = (const float*)d_in[3];   // [D,3D]
  const float* Wp   = (const float*)d_in[4];   // [D,D]
  float* out = (float*)d_out;                  // [B,N,D] fp32

  // workspace: Q/K/V [B,H,N,DH] + O [B,N,D], all bf16 (u16) -> 64 MB total
  unsigned short* Qb = (unsigned short*)d_ws;
  const size_t perBuf = (size_t)Bc * Hc * Nc * DHc;   // 8,388,608 elements
  unsigned short* Kb = Qb + perBuf;
  unsigned short* Vb = Kb + perBuf;
  unsigned short* Ob = Vb + perBuf;

  dim3 g1(Bc * Nc / 64, (3 * Dc) / 64);   // 256 x 24
  qkv_kernel<<<g1, 256, 0, stream>>>(x, Wqkv, Qb, Kb, Vb);

  dim3 g2(Nc / 128, Hc, Bc);              // 16 x 8 x 8
  attn_kernel<<<g2, 256, 0, stream>>>(Qb, Kb, Vb, bias, mask, Ob);

  dim3 g3(Bc * Nc / 64, Dc / 64);         // 256 x 8
  proj_kernel<<<g3, 256, 0, stream>>>(Ob, Wp, out);
}
